// WorkingMemory_40647570489696
// MI455X (gfx1250) — matrix operand order repaired and checked by reading, not compiled, not run on hardware
//
#include <hip/hip_runtime.h>
#include <stdint.h>

// ---------------------------------------------------------------------------
// LSTM for MI455X (gfx1250, wave32).
//   Phase 1: x_proj = inputs @ w_ih^T + (b_ih + b_hh)   [16384 x 4096] fp32
//            via v_wmma_f32_16x16x32_bf16 (bf16 in, fp32 acc).
//   Phase 2: persistent scan kernel, 64 WGs, grid barrier per timestep.
//            w_hh (8MB bf16) is L2-resident (192MB L2); c-state in registers.
//            Dual accumulators break the per-step WMMA RAW chain; x_proj
//            slice is prefetched ahead of the matmul to hide HBM latency.
// ---------------------------------------------------------------------------

typedef __attribute__((ext_vector_type(16))) __bf16 v16bf;
typedef __attribute__((ext_vector_type(8)))  __bf16 v8bf;
typedef __attribute__((ext_vector_type(8)))  float  v8f;
typedef __attribute__((ext_vector_type(4)))  float  f32x4;
typedef __attribute__((ext_vector_type(4)))  unsigned short u16x4;

union Frag { v16bf v; v8bf h[2]; };

#define LSTM_B   32
#define LSTM_T   512
#define LSTM_I   1024
#define LSTM_H   1024
#define LSTM_G   4096          // 4*H
#define LSTM_BT  16384         // B*T
#define NWG      64            // persistent workgroups in scan kernel

static __device__ __forceinline__ unsigned short f32_to_bf16_rne(float f) {
    unsigned int u = __float_as_uint(f);
    u = (u + 0x7FFFu + ((u >> 16) & 1u)) >> 16;
    return (unsigned short)u;
}

// ---------------- fp32 -> bf16 convert (vectorized x4) ---------------------
__global__ __launch_bounds__(256) void cvt_f32_bf16(const float* __restrict__ s,
                                                    unsigned short* __restrict__ d,
                                                    int n) {
    int i = (blockIdx.x * 256 + threadIdx.x) * 4;
    if (i < n) {
        f32x4 f = *(const f32x4*)(s + i);
        u16x4 u;
        u.x = f32_to_bf16_rne(f.x);
        u.y = f32_to_bf16_rne(f.y);
        u.z = f32_to_bf16_rne(f.z);
        u.w = f32_to_bf16_rne(f.w);
        *(u16x4*)(d + i) = u;
    }
}

// ---------------- zero h-state + per-step counters (replay safe) -----------
__global__ __launch_bounds__(256) void init_state(unsigned int* __restrict__ hbf_w,
                                                  unsigned int* __restrict__ ctr) {
    int i = blockIdx.x * 256 + threadIdx.x;
    if (i < (LSTM_B * LSTM_H / 2)) hbf_w[i] = 0u;   // 32768 bf16 = 16384 dwords
    if (i < LSTM_T) ctr[i] = 0u;
}

// ---------------- Phase 1: x_proj GEMM -------------------------------------
// A = inputs_bf16 [16384,1024] row-major, B = w_ih_bf16 [4096,1024] row-major
// (B[k][n] = w[n][k] -> contiguous-K loads, no transpose).
// Workgroup = 8 waves in 2(M) x 4(N); each wave -> 32x32 (four 16x16 tiles).
__global__ __launch_bounds__(256) void gemm_xproj(
    const unsigned short* __restrict__ A,
    const unsigned short* __restrict__ Bw,
    const float* __restrict__ b_ih,
    const float* __restrict__ b_hh,
    float* __restrict__ Xp) {
    const int lane = threadIdx.x & 31;
    const int wave = threadIdx.x >> 5;
    const int wm = wave >> 2;                     // 0..1
    const int wn = wave & 3;                      // 0..3
    const int m0 = blockIdx.y * 64 + wm * 32;
    const int n0 = blockIdx.x * 128 + wn * 32;

    const int lm   = lane & 15;
    const int lk8  = (lane >> 4) * 8;             // A-frag chunk select
    const int lk16 = (lane >> 4) * 16;            // unused; B-frags also use lk8

    const unsigned short* arow0 = A  + (size_t)(m0 + lm)      * LSTM_I;
    const unsigned short* arow1 = A  + (size_t)(m0 + 16 + lm) * LSTM_I;
    const unsigned short* brow0 = Bw + (size_t)(n0 + lm)      * LSTM_I;
    const unsigned short* brow1 = Bw + (size_t)(n0 + 16 + lm) * LSTM_I;

    v8f acc00 = {}, acc01 = {}, acc10 = {}, acc11 = {};

    for (int k0 = 0; k0 < LSTM_I; k0 += 32) {
        Frag a0, a1, b0, b1;
        a0.h[0] = *(const v8bf*)(arow0 + k0 + lk8);
        a0.h[1] = *(const v8bf*)(arow0 + k0 + 16 + lk8);
        a1.h[0] = *(const v8bf*)(arow1 + k0 + lk8);
        a1.h[1] = *(const v8bf*)(arow1 + k0 + 16 + lk8);
        b0.h[0] = *(const v8bf*)(brow0 + k0 + lk8);
        b0.h[1] = *(const v8bf*)(brow0 + k0 + 16 + lk8);
        b1.h[0] = *(const v8bf*)(brow1 + k0 + lk8);
        b1.h[1] = *(const v8bf*)(brow1 + k0 + 16 + lk8);
        acc00 = __builtin_amdgcn_wmma_f32_16x16x32_bf16(false, a0.v, false, b0.v, (short)0, acc00, false, false);
        acc01 = __builtin_amdgcn_wmma_f32_16x16x32_bf16(false, a0.v, false, b1.v, (short)0, acc01, false, false);
        acc10 = __builtin_amdgcn_wmma_f32_16x16x32_bf16(false, a1.v, false, b0.v, (short)0, acc10, false, false);
        acc11 = __builtin_amdgcn_wmma_f32_16x16x32_bf16(false, a1.v, false, b1.v, (short)0, acc11, false, false);
    }

    // C/D layout: n = lane&15, m = (lane>>4)*8 + p
    const int cn = lane & 15;
    const int cm = (lane >> 4) * 8;
    const int nA = n0 + cn;
    const int nB = n0 + 16 + cn;
    const float biasA = b_ih[nA] + b_hh[nA];
    const float biasB = b_ih[nB] + b_hh[nB];
#pragma unroll
    for (int p = 0; p < 8; ++p) {
        const size_t mA = (size_t)(m0 + cm + p);
        const size_t mB = (size_t)(m0 + 16 + cm + p);
        Xp[mA * LSTM_G + nA] = acc00[p] + biasA;
        Xp[mA * LSTM_G + nB] = acc01[p] + biasB;
        Xp[mB * LSTM_G + nA] = acc10[p] + biasA;
        Xp[mB * LSTM_G + nB] = acc11[p] + biasB;
    }
}

// ---------------- Phase 2: persistent recurrent scan -----------------------
// 64 WGs x 256 thr. WG owns hidden columns [j0, j0+16). Wave (mi,gate):
// 16x16 tile: rows = batch mi*16..+15, cols = gate*H + j0..+15. K-loop over h.
__global__ __launch_bounds__(256) void lstm_scan(
    const float* __restrict__ Xp,            // [B*T, 4H] fp32 (biases folded)
    const unsigned short* __restrict__ Whh,  // [4H, H] bf16
    unsigned short* __restrict__ Hbf,        // [B, H] bf16 (recurrent state)
    float* __restrict__ out,                 // outputs | h_f | c_f
    unsigned int* __restrict__ ctr) {        // [T] arrive counters (zeroed)
    __shared__ float gl[4][LSTM_B][16];      // activated gate tiles (8 KB)

    const int tid  = threadIdx.x;
    const int lane = tid & 31;
    const int wave = tid >> 5;
    const int mi   = wave >> 2;              // 0..1  (batch half)
    const int gate = wave & 3;               // i,f,g,o
    const int j0   = blockIdx.x * 16;

    const int lm   = lane & 15;
    const int lk8  = (lane >> 4) * 8;
    const int lk16 = (lane >> 4) * 16;

    const unsigned short* arow = Hbf + (size_t)(mi * 16 + lm) * LSTM_H;
    const unsigned short* brow = Whh + (size_t)(gate * LSTM_H + j0 + lm) * LSTM_H;

    const int cn = lane & 15;
    const int cm = (lane >> 4) * 8;
    const int ncol = gate * LSTM_H + j0 + cn;

    // combine-phase ownership: element e in [0,512): b = e>>4, j = e&15
    const int b0e = tid >> 4;                 // e0 = tid
    const int b1e = (tid + 256) >> 4;         // e1 = tid + 256
    const int je  = tid & 15;
    float c0 = 0.f, c1 = 0.f;

    for (int t = 0; t < LSTM_T; ++t) {
        // Prefetch this step's x_proj slice (HBM-resident) so the epilogue
        // loads hit cache; latency hidden behind the whole K-loop.
#pragma unroll
        for (int p = 0; p < 8; ++p) {
            const int b = mi * 16 + cm + p;
            __builtin_prefetch(&Xp[((size_t)b * LSTM_T + t) * LSTM_G + ncol], 0, 0);
        }

        // Dual accumulators: two independent WMMA chains (even/odd K blocks).
        v8f accE = {}, accO = {};
        for (int k0 = 0; k0 < LSTM_H; k0 += 64) {
            Frag a0, b0, a1, b1;
            a0.h[0] = *(const v8bf*)(arow + k0 + lk8);
            a0.h[1] = *(const v8bf*)(arow + k0 + 16 + lk8);
            b0.h[0] = *(const v8bf*)(brow + k0 + lk8);
            b0.h[1] = *(const v8bf*)(brow + k0 + 16 + lk8);
            a1.h[0] = *(const v8bf*)(arow + k0 + 32 + lk8);
            a1.h[1] = *(const v8bf*)(arow + k0 + 48 + lk8);
            b1.h[0] = *(const v8bf*)(brow + k0 + 32 + lk8);
            b1.h[1] = *(const v8bf*)(brow + k0 + 48 + lk8);
            accE = __builtin_amdgcn_wmma_f32_16x16x32_bf16(false, a0.v, false, b0.v, (short)0, accE, false, false);
            accO = __builtin_amdgcn_wmma_f32_16x16x32_bf16(false, a1.v, false, b1.v, (short)0, accO, false, false);
        }
        v8f acc = accE + accO;

        // add x_proj slice, activate, stage in LDS
#pragma unroll
        for (int p = 0; p < 8; ++p) {
            const int b = mi * 16 + cm + p;
            float g = acc[p] + Xp[((size_t)b * LSTM_T + t) * LSTM_G + ncol];
            g = (gate == 2) ? tanhf(g) : 1.f / (1.f + __expf(-g));
            gl[gate][b][cn] = g;
        }
        __syncthreads();
        // combine i,f,g,o -> c,h ; c kept in registers across all steps
        {
            float iG = gl[0][b0e][je], fG = gl[1][b0e][je];
            float gG = gl[2][b0e][je], oG = gl[3][b0e][je];
            c0 = fG * c0 + iG * gG;
            float h = oG * tanhf(c0);
            out[((size_t)b0e * LSTM_T + t) * LSTM_H + j0 + je] = h;
            Hbf[b0e * LSTM_H + j0 + je] = f32_to_bf16_rne(h);
            if (t == LSTM_T - 1) {
                out[(size_t)LSTM_BT * LSTM_H + b0e * LSTM_H + j0 + je] = h;
                out[(size_t)LSTM_BT * LSTM_H + LSTM_B * LSTM_H + b0e * LSTM_H + j0 + je] = c0;
            }
        }
        {
            float iG = gl[0][b1e][je], fG = gl[1][b1e][je];
            float gG = gl[2][b1e][je], oG = gl[3][b1e][je];
            c1 = fG * c1 + iG * gG;
            float h = oG * tanhf(c1);
            out[((size_t)b1e * LSTM_T + t) * LSTM_H + j0 + je] = h;
            Hbf[b1e * LSTM_H + j0 + je] = f32_to_bf16_rne(h);
            if (t == LSTM_T - 1) {
                out[(size_t)LSTM_BT * LSTM_H + b1e * LSTM_H + j0 + je] = h;
                out[(size_t)LSTM_BT * LSTM_H + LSTM_B * LSTM_H + b1e * LSTM_H + j0 + je] = c1;
            }
        }
        // device-wide barrier for step t (release h, arrive, acquire)
        __threadfence();
        __syncthreads();
        if (tid == 0) {
            atomicAdd(&ctr[t], 1u);
            while (__hip_atomic_load(&ctr[t], __ATOMIC_RELAXED, __HIP_MEMORY_SCOPE_AGENT) < (unsigned)NWG)
                __builtin_amdgcn_s_sleep(2);
        }
        __syncthreads();
        __threadfence();
    }
}

// ---------------------------------------------------------------------------
extern "C" void kernel_launch(void* const* d_in, const int* in_sizes, int n_in,
                              void* d_out, int out_size, void* d_ws, size_t ws_size,
                              hipStream_t stream) {
    const float* inputs = (const float*)d_in[0];   // [B,T,I]
    const float* w_ih   = (const float*)d_in[1];   // [4H,I]
    const float* w_hh   = (const float*)d_in[2];   // [4H,H]
    const float* b_ih   = (const float*)d_in[3];   // [4H]
    const float* b_hh   = (const float*)d_in[4];   // [4H]
    float* out = (float*)d_out;

    char* ws = (char*)d_ws;
    const size_t XPROJ_B = (size_t)LSTM_BT * LSTM_G * sizeof(float);     // 256 MB
    const size_t INBF_B  = (size_t)LSTM_BT * LSTM_I * sizeof(unsigned short);
    const size_t WIH_B   = (size_t)LSTM_G  * LSTM_I * sizeof(unsigned short);
    const size_t WHH_B   = (size_t)LSTM_G  * LSTM_H * sizeof(unsigned short);
    const size_t HBF_B   = (size_t)LSTM_B  * LSTM_H * sizeof(unsigned short);

    float*          x_proj = (float*)ws;
    unsigned short* in_bf  = (unsigned short*)(ws + XPROJ_B);
    unsigned short* wih_bf = (unsigned short*)(ws + XPROJ_B + INBF_B);
    unsigned short* whh_bf = (unsigned short*)(ws + XPROJ_B + INBF_B + WIH_B);
    unsigned short* h_bf   = (unsigned short*)(ws + XPROJ_B + INBF_B + WIH_B + WHH_B);
    unsigned int*   ctr    = (unsigned int*)(ws + XPROJ_B + INBF_B + WIH_B + WHH_B + HBF_B);

    // fp32 -> bf16 conversions
    cvt_f32_bf16<<<(LSTM_BT * LSTM_I / 4 + 255) / 256, 256, 0, stream>>>(inputs, in_bf, LSTM_BT * LSTM_I);
    cvt_f32_bf16<<<(LSTM_G * LSTM_I / 4 + 255) / 256, 256, 0, stream>>>(w_ih, wih_bf, LSTM_G * LSTM_I);
    cvt_f32_bf16<<<(LSTM_G * LSTM_H / 4 + 255) / 256, 256, 0, stream>>>(w_hh, whh_bf, LSTM_G * LSTM_H);

    // zero h0 and per-step counters (must re-run every launch for graph replay)
    init_state<<<64, 256, 0, stream>>>((unsigned int*)h_bf, ctr);

    // Phase 1: big WMMA GEMM
    gemm_xproj<<<dim3(LSTM_G / 128, LSTM_BT / 64), 256, 0, stream>>>(in_bf, wih_bf, b_ih, b_hh, x_proj);

    // Phase 2: persistent scan
    lstm_scan<<<NWG, 256, 0, stream>>>(x_proj, whh_bf, h_bf, out, ctr);
}
